// MultiHeadAttention_55972013801989
// MI455X (gfx1250) — matrix operand order repaired and checked by reading, not compiled, not run on hardware
//
#include <hip/hip_runtime.h>
#include <hip/hip_bf16.h>

// ---- Problem constants (match reference) ----
#define B_SZ   4
#define T_SEQ  2048
#define DM     1024
#define NH     16
#define DK     64               // DM / NH
#define MTOT   (B_SZ * T_SEQ)   // 8192 rows in token-major GEMMs

typedef _Float16 v16h __attribute__((ext_vector_type(16)));
typedef float    v8f  __attribute__((ext_vector_type(8)));
typedef _Float16 h4   __attribute__((ext_vector_type(4)));

union FragA {
    v16h v;
    uint4 q[2];
};

__device__ __forceinline__ v8f wmma_f16(v16h a, v16h b, v8f c) {
    // D = A(16x32 f16) * B(32x16 f16) + C(16x16 f32)
    return __builtin_amdgcn_wmma_f32_16x16x32_f16(
        /*neg_a=*/false, a, /*neg_b=*/false, b,
        /*c_mod=*/(short)0, c, /*reuse_a=*/false, /*reuse_b=*/false);
}

// ---- gfx1250 async global->LDS path (GLOBAL_LOAD_ASYNC_TO_LDS_B128, ASYNCcnt) ----
#if defined(__HIP_DEVICE_COMPILE__) && __has_builtin(__builtin_amdgcn_global_load_async_to_lds_b128)
#define USE_ASYNC_LDS 1
#else
#define USE_ASYNC_LDS 0
#endif

// Pointer types matching the builtin's signature:
//   param0: int __vector(4) __device__ *   (addrspace(1), GCC-style vector)
//   param1: int __vector(4) __shared__ *   (addrspace(3))
typedef int b128i __attribute__((vector_size(16)));
typedef __attribute__((address_space(1))) b128i* gas_b128;
typedef __attribute__((address_space(3))) b128i* las_b128;

// 16B memory -> LDS copy (async when available; sync fallback keeps compile safe)
__device__ __forceinline__ void cp_b128_to_lds(_Float16* lds_dst, const _Float16* gsrc) {
#if USE_ASYNC_LDS
    __builtin_amdgcn_global_load_async_to_lds_b128(
        (gas_b128)gsrc, (las_b128)lds_dst,
        /*imm offset=*/0, /*cpol=*/0);
#else
    *(uint4*)lds_dst = *(const uint4*)gsrc;
#endif
}

__device__ __forceinline__ void wait_async_all() {
#if defined(__HIP_DEVICE_COMPILE__)
#if __has_builtin(__builtin_amdgcn_s_wait_asynccnt)
    __builtin_amdgcn_s_wait_asynccnt(0);
#else
    asm volatile("s_wait_asynccnt 0x0" ::: "memory");
#endif
#endif
}

// -------------------------------------------------------------------------
// f32 -> f16 conversion (x), 4 elems / thread, vectorized
// -------------------------------------------------------------------------
__global__ __launch_bounds__(256) void f32_to_f16_kernel(
    const float* __restrict__ in, _Float16* __restrict__ out, int n4)
{
    int i = blockIdx.x * 256 + threadIdx.x;
    if (i < n4) {
        float4 v = ((const float4*)in)[i];
        h4 o = { (_Float16)v.x, (_Float16)v.y, (_Float16)v.z, (_Float16)v.w };
        ((h4*)out)[i] = o;
    }
}

// -------------------------------------------------------------------------
// W[k][n] f32 -> Wt[n][k] f16  (1024x1024), 32x32 LDS tiles
// -------------------------------------------------------------------------
__global__ __launch_bounds__(256) void transpose_f16_kernel(
    const float* __restrict__ W, _Float16* __restrict__ Wt)
{
    __shared__ float tile[32][33];
    const int tx = threadIdx.x, ty = threadIdx.y;   // 32 x 8
    const int n0 = blockIdx.x * 32, k0 = blockIdx.y * 32;
    #pragma unroll
    for (int j = 0; j < 32; j += 8)
        tile[ty + j][tx] = W[(size_t)(k0 + ty + j) * DM + n0 + tx];
    __syncthreads();
    #pragma unroll
    for (int j = 0; j < 32; j += 8)
        Wt[(size_t)(n0 + ty + j) * DM + k0 + tx] = (_Float16)tile[tx][ty + j];
}

// -------------------------------------------------------------------------
// Tiled WMMA GEMM:  C[M,N] = A[M,K] * Bt[N,K]^T + bias
//   M = 8192, N = K = 1024.  Block tile 128x128, K-step 32.
//   Double-buffered LDS; tiles staged with async global->LDS (prefetch tile
//   i+1 while WMMAs consume tile i).
//   8 waves: wave grid 2(M) x 4(N); per wave 4x2 WMMA tiles (64x32).
//   MODE 0: f16 row-major out (+bias)       (Q, K projections)
//   MODE 1: f16 per-head-transposed (+bias) (V projection -> Vt[b,h,d,t])
//   MODE 2: f32 row-major out (+bias)       (output projection)
// -------------------------------------------------------------------------
template<int MODE>
__global__ __launch_bounds__(256) void gemm128_wmma(
    const _Float16* __restrict__ A,
    const _Float16* __restrict__ Bt,
    const float*    __restrict__ bias,
    void* __restrict__ out)
{
    constexpr int K   = DM;
    constexpr int N   = DM;
    constexpr int LDT = 40;   // padded LDS stride (halves): 80B -> conflict-free b128 reads
    __shared__ _Float16 As[2][128 * LDT];
    __shared__ _Float16 Bs[2][128 * LDT];

    const int tid  = threadIdx.x;
    const int w    = tid >> 5, lane = tid & 31;
    const int half = lane >> 4, l16 = lane & 15;
    const int wm   = w & 1, wn = w >> 1;
    const int m0   = blockIdx.y * 128;
    const int n0   = blockIdx.x * 128;

    // per-thread tile-load assignment: 512 x 16B chunks per matrix, 2 per thread
    const int li0  = tid * 2;
    const int lrow0 = li0 >> 2,       lc0 = (li0 & 3) * 8;
    const int lrow1 = (li0 + 1) >> 2, lc1 = ((li0 + 1) & 3) * 8;

    auto load_tile = [&](int buf, int k0) {
        cp_b128_to_lds(&As[buf][lrow0 * LDT + lc0], A  + (size_t)(m0 + lrow0) * K + k0 + lc0);
        cp_b128_to_lds(&Bs[buf][lrow0 * LDT + lc0], Bt + (size_t)(n0 + lrow0) * K + k0 + lc0);
        cp_b128_to_lds(&As[buf][lrow1 * LDT + lc1], A  + (size_t)(m0 + lrow1) * K + k0 + lc1);
        cp_b128_to_lds(&Bs[buf][lrow1 * LDT + lc1], Bt + (size_t)(n0 + lrow1) * K + k0 + lc1);
    };

    v8f acc[4][2];
    #pragma unroll
    for (int mt = 0; mt < 4; ++mt)
        #pragma unroll
        for (int nt = 0; nt < 2; ++nt) acc[mt][nt] = {};

    load_tile(0, 0);
    int buf = 0;

    for (int k0 = 0; k0 < K; k0 += 32) {
        wait_async_all();        // tile `buf` landed in LDS (this wave's loads)
        __syncthreads();         // ... and everyone else's; prior reads retired
        if (k0 + 32 < K) load_tile(buf ^ 1, k0 + 32);   // overlap next tile

        FragA af[4], bf[2];
        #pragma unroll
        for (int mt = 0; mt < 4; ++mt) {        // A: K = e%8 + (e/8)*16 + half*8
            int row = wm * 64 + mt * 16 + l16;
            af[mt].q[0] = *(const uint4*)&As[buf][row * LDT + half * 8];
            af[mt].q[1] = *(const uint4*)&As[buf][row * LDT + half * 8 + 16];
        }
        #pragma unroll
        for (int nt = 0; nt < 2; ++nt) {        // B: K = e%8 + (e/8)*16 + half*8, N = l16
            int nr = wn * 32 + nt * 16 + l16;
            bf[nt].q[0] = *(const uint4*)&Bs[buf][nr * LDT + half * 8];
            bf[nt].q[1] = *(const uint4*)&Bs[buf][nr * LDT + half * 8 + 16];
        }
        #pragma unroll
        for (int mt = 0; mt < 4; ++mt)
            #pragma unroll
            for (int nt = 0; nt < 2; ++nt)
                acc[mt][nt] = wmma_f16(af[mt].v, bf[nt].v, acc[mt][nt]);

        buf ^= 1;
    }

    // Epilogue: C layout lane mapping: M = r + half*8, N = l16
    #pragma unroll
    for (int nt = 0; nt < 2; ++nt) {
        const int col = n0 + wn * 32 + nt * 16 + l16;
        const float bv = bias[col];
        #pragma unroll
        for (int mt = 0; mt < 4; ++mt) {
            #pragma unroll
            for (int r = 0; r < 8; ++r) {
                const int row = m0 + wm * 64 + mt * 16 + r + half * 8;
                const float v = acc[mt][nt][r] + bv;
                if (MODE == 0) {
                    ((_Float16*)out)[(size_t)row * N + col] = (_Float16)v;
                } else if (MODE == 1) {
                    const int bb = row / T_SEQ, t = row % T_SEQ;
                    const int hh = col / DK,   d = col % DK;
                    ((_Float16*)out)[((size_t)((bb * NH + hh) * DK + d)) * T_SEQ + t] =
                        (_Float16)v;
                } else {
                    ((float*)out)[(size_t)row * N + col] = v;
                }
            }
        }
    }
}

// -------------------------------------------------------------------------
// Flash attention (causal). Block = (b, h, 128 q-rows); wave = 16 q-rows.
// Q[b*T+t, h*64+d] f16, K same layout, Vt[((b*NH+h)*DK+d)*T + t] f16.
// Output Oh token-major f16 [MTOT, DM].
// Mask input is exactly tril -> applied analytically (t <= q).
// -------------------------------------------------------------------------
__global__ __launch_bounds__(256) void flash_attn_kernel(
    const _Float16* __restrict__ Q,
    const _Float16* __restrict__ Kh,
    const _Float16* __restrict__ Vt,
    _Float16* __restrict__ Oh)
{
    __shared__ _Float16 Ps[8 * 16 * 32];    // per-wave 16x32 P staging (C->A relayout)

    const int tid  = threadIdx.x;
    const int w    = tid >> 5, lane = tid & 31;
    const int half = lane >> 4, l16 = lane & 15;
    const int nq   = T_SEQ / 128;
    const int bid  = blockIdx.x;
    const int b    = bid / (NH * nq);
    const int h    = (bid / nq) % NH;
    const int qb   = (bid % nq) * 128;
    const int q0   = qb + w * 16;

    // Q fragments, reused across the whole key loop (A layout, kk = 0 and 32)
    const _Float16* qrow = Q + ((size_t)(b * T_SEQ + q0 + l16)) * DM + h * DK;
    FragA aq0, aq1;
    aq0.q[0] = *(const uint4*)(qrow + half * 8);
    aq0.q[1] = *(const uint4*)(qrow + half * 8 + 16);
    aq1.q[0] = *(const uint4*)(qrow + 32 + half * 8);
    aq1.q[1] = *(const uint4*)(qrow + 32 + half * 8 + 16);

    float m_i[8], l_i[8];
    v8f acc[4];
    #pragma unroll
    for (int r = 0; r < 8; ++r) { m_i[r] = -3.0e38f; l_i[r] = 0.0f; }
    #pragma unroll
    for (int dt = 0; dt < 4; ++dt) acc[dt] = {};

    _Float16* ps = &Ps[w * 512];
    const int q_last = q0 + 15;

    for (int tb = 0; tb <= q_last; tb += 32) {
        // ---- scores: two 16x16 tiles over t in [tb, tb+32) ----
        v8f s0 = {}, s1 = {};
        {
            const _Float16* krow = Kh + ((size_t)(b * T_SEQ + tb + l16)) * DM
                                      + h * DK + half * 8;
            FragA bk0, bk1;
            bk0.q[0] = *(const uint4*)(krow);
            bk0.q[1] = *(const uint4*)(krow + 16);
            bk1.q[0] = *(const uint4*)(krow + 32);
            bk1.q[1] = *(const uint4*)(krow + 48);
            s0 = wmma_f16(aq0.v, bk0.v, s0);
            s0 = wmma_f16(aq1.v, bk1.v, s0);

            const _Float16* krow2 = krow + (size_t)16 * DM;
            FragA ck0, ck1;
            ck0.q[0] = *(const uint4*)(krow2);
            ck0.q[1] = *(const uint4*)(krow2 + 16);
            ck1.q[0] = *(const uint4*)(krow2 + 32);
            ck1.q[1] = *(const uint4*)(krow2 + 48);
            s1 = wmma_f16(aq0.v, ck0.v, s1);
            s1 = wmma_f16(aq1.v, ck1.v, s1);

            // prefetch next key tile while softmax runs (global_prefetch_b8)
            if (tb + 32 <= q_last) {
                __builtin_prefetch(krow  + (size_t)32 * DM, 0, 3);
                __builtin_prefetch(krow2 + (size_t)32 * DM, 0, 3);
            }
        }

        // ---- scale + causal mask (+ -1e9 like the reference) ----
        const int t0 = tb + l16, t1 = tb + 16 + l16;
        float p0[8], p1[8], mt[8];
        #pragma unroll
        for (int r = 0; r < 8; ++r) {
            const int qg = q0 + r + half * 8;
            float v0 = s0[r] * 0.125f; if (t0 > qg) v0 = -1.0e9f;
            float v1 = s1[r] * 0.125f; if (t1 > qg) v1 = -1.0e9f;
            p0[r] = v0; p1[r] = v1;
            mt[r] = fmaxf(v0, v1);
        }
        // row-max across the 16 lanes of each half-wave (rows are per-half)
        #pragma unroll
        for (int off = 8; off >= 1; off >>= 1)
            #pragma unroll
            for (int r = 0; r < 8; ++r)
                mt[r] = fmaxf(mt[r], __shfl_xor(mt[r], off, 16));

        float corr[8];
        #pragma unroll
        for (int r = 0; r < 8; ++r) {
            const float mn = fmaxf(m_i[r], mt[r]);
            corr[r] = __expf(m_i[r] - mn);
            m_i[r] = mn;
            p0[r] = __expf(p0[r] - mn);
            p1[r] = __expf(p1[r] - mn);
        }
        float rs[8];
        #pragma unroll
        for (int r = 0; r < 8; ++r) rs[r] = p0[r] + p1[r];
        #pragma unroll
        for (int off = 8; off >= 1; off >>= 1)
            #pragma unroll
            for (int r = 0; r < 8; ++r)
                rs[r] += __shfl_xor(rs[r], off, 16);
        #pragma unroll
        for (int r = 0; r < 8; ++r) l_i[r] = l_i[r] * corr[r] + rs[r];
        #pragma unroll
        for (int dt = 0; dt < 4; ++dt)
            #pragma unroll
            for (int r = 0; r < 8; ++r) acc[dt][r] *= corr[r];

        // ---- P: C layout -> LDS -> A layout ----
        #pragma unroll
        for (int r = 0; r < 8; ++r) {
            ps[(r + half * 8) * 32 + l16]      = (_Float16)p0[r];
            ps[(r + half * 8) * 32 + 16 + l16] = (_Float16)p1[r];
        }
        asm volatile("s_wait_dscnt 0" ::: "memory");  // intra-wave LDS RAW fence
        FragA ap;
        ap.q[0] = *(const uint4*)&ps[l16 * 32 + half * 8];
        ap.q[1] = *(const uint4*)&ps[l16 * 32 + half * 8 + 16];

        // ---- P(16x32) * V(32x64): 4 WMMAs; Vt rows are t-contiguous ----
        #pragma unroll
        for (int dt = 0; dt < 4; ++dt) {
            const _Float16* vrow = Vt
                + ((size_t)((b * NH + h) * DK + dt * 16 + l16)) * T_SEQ
                + tb + half * 8;
            FragA bv;
            bv.q[0] = *(const uint4*)(vrow);
            bv.q[1] = *(const uint4*)(vrow + 16);
            acc[dt] = wmma_f16(ap.v, bv.v, acc[dt]);
        }
    }

    // ---- normalize and store (token-major f16, head columns) ----
    #pragma unroll
    for (int dt = 0; dt < 4; ++dt)
        #pragma unroll
        for (int r = 0; r < 8; ++r) {
            const int qg = q0 + r + half * 8;
            Oh[((size_t)(b * T_SEQ + qg)) * DM + h * DK + dt * 16 + l16] =
                (_Float16)(acc[dt][r] / l_i[r]);
        }
}

// -------------------------------------------------------------------------
// Host-side launch
// -------------------------------------------------------------------------
extern "C" void kernel_launch(void* const* d_in, const int* in_sizes, int n_in,
                              void* d_out, int out_size, void* d_ws, size_t ws_size,
                              hipStream_t stream)
{
    (void)in_sizes; (void)n_in; (void)out_size; (void)ws_size;
    const float* x   = (const float*)d_in[0];
    // d_in[1] = causal tril mask; applied analytically in flash_attn_kernel
    const float* w_q = (const float*)d_in[2];
    const float* b_q = (const float*)d_in[3];
    const float* w_k = (const float*)d_in[4];
    const float* b_k = (const float*)d_in[5];
    const float* w_v = (const float*)d_in[6];
    const float* b_v = (const float*)d_in[7];
    const float* w_o = (const float*)d_in[8];
    const float* b_o = (const float*)d_in[9];

    char* ws = (char*)d_ws;
    const size_t SZ_X = (size_t)MTOT * DM * 2;   // 16 MB
    const size_t SZ_W = (size_t)DM * DM * 2;     // 2 MB
    _Float16* xh    = (_Float16*)(ws);
    _Float16* wqT   = (_Float16*)(ws + SZ_X);
    _Float16* wkT   = (_Float16*)(ws + SZ_X + SZ_W);
    _Float16* wvT   = (_Float16*)(ws + SZ_X + 2 * SZ_W);
    _Float16* woT   = (_Float16*)(ws + SZ_X + 3 * SZ_W);
    _Float16* Qh    = (_Float16*)(ws + SZ_X + 4 * SZ_W);
    _Float16* Kh    = (_Float16*)(ws + 2 * SZ_X + 4 * SZ_W);
    _Float16* Vth   = (_Float16*)(ws + 3 * SZ_X + 4 * SZ_W);
    _Float16* attnh = (_Float16*)(ws + 4 * SZ_X + 4 * SZ_W);

    // 1) conversions
    const int n4 = MTOT * DM / 4;
    f32_to_f16_kernel<<<(n4 + 255) / 256, 256, 0, stream>>>(x, xh, n4);
    dim3 tb(32, 8), tg(DM / 32, DM / 32);
    transpose_f16_kernel<<<tg, tb, 0, stream>>>(w_q, wqT);
    transpose_f16_kernel<<<tg, tb, 0, stream>>>(w_k, wkT);
    transpose_f16_kernel<<<tg, tb, 0, stream>>>(w_v, wvT);
    transpose_f16_kernel<<<tg, tb, 0, stream>>>(w_o, woT);

    // 2) Q/K/V projections (WMMA GEMM, async double-buffered LDS)
    dim3 gg(DM / 128, MTOT / 128);   // (8, 64)
    gemm128_wmma<0><<<gg, 256, 0, stream>>>(xh, wqT, b_q, (void*)Qh);
    gemm128_wmma<0><<<gg, 256, 0, stream>>>(xh, wkT, b_k, (void*)Kh);
    gemm128_wmma<1><<<gg, 256, 0, stream>>>(xh, wvT, b_v, (void*)Vth);

    // 3) causal flash attention
    flash_attn_kernel<<<B_SZ * NH * (T_SEQ / 128), 256, 0, stream>>>(Qh, Kh, Vth, attnh);

    // 4) output projection -> f32 d_out
    gemm128_wmma<2><<<gg, 256, 0, stream>>>(attnh, woT, b_o, d_out);
}
